// Decoder_73486890434953
// MI455X (gfx1250) — compile-verified, matrix operand order checked by reading, not run on hardware
//
#include <hip/hip_runtime.h>
#include <hip/hip_bf16.h>
#include <stdint.h>

// Problem constants (match reference)
#define Bn     32
#define Tn     512
#define Hn     256
#define Pn     16
#define NITERS 4
#define Mrows  (Bn * Tn)     // 16384 score rows
#define TWOH   (2 * Hn)      // 512
#define THREEH (3 * Hn)      // 768
#define FOURH  (4 * Hn)      // 1024
#define FIVEH  (5 * Hn)      // 1280
#define HP     (Hn * Pn)     // 4096
#define MASK_PENALTY 1.0e9f

typedef __attribute__((ext_vector_type(16))) __bf16 v16bf;
typedef __attribute__((ext_vector_type(8)))  float  v8f;
typedef unsigned int u32x4 __attribute__((ext_vector_type(4)));
typedef int          i32x4 __attribute__((ext_vector_type(4)));
typedef int          i32x8 __attribute__((ext_vector_type(8)));

// Tensor Data Mover availability (device pass only; host pass uses fallback,
// its kernel bodies are discarded anyway).
#if defined(__HIP_DEVICE_COMPILE__) && \
    __has_builtin(__builtin_amdgcn_tensor_load_to_lds) && \
    __has_builtin(__builtin_amdgcn_s_wait_tensorcnt)
#define ATHENA_USE_TDM 1
#else
#define ATHENA_USE_TDM 0
#endif

static __device__ __forceinline__ unsigned short f2bf(float f) {
  unsigned int u = __builtin_bit_cast(unsigned int, f);
  u += 0x7FFFu + ((u >> 16) & 1u);   // round-to-nearest-even
  return (unsigned short)(u >> 16);
}

// ---------------------------------------------------------------------------
// One-time conversions
// ---------------------------------------------------------------------------
__global__ void k_f32_to_bf16(const float* __restrict__ in,
                              unsigned short* __restrict__ out, int n) {
  int i = blockIdx.x * blockDim.x + threadIdx.x;
  if (i < n) out[i] = f2bf(in[i]);
}

// er[:, :, 0:512] = Et (transpose of encoding_matrix), bf16
__global__ void k_build_er_base(const float* __restrict__ enc,
                                unsigned short* __restrict__ er) {
  int idx = blockIdx.x * blockDim.x + threadIdx.x;   // over Mrows * 2H
  if (idx >= Mrows * TWOH) return;
  int m = idx / TWOH, c = idx % TWOH;
  int b = m / Tn, t = m % Tn;
  er[(size_t)m * THREEH + c] = f2bf(enc[((size_t)b * TWOH + c) * Tn + t]);
}

// hx = cx = 0 ; enc_s = enc_e = Et[b, 0]
__global__ void k_init_state(const float* __restrict__ enc,
                             float* __restrict__ hx, float* __restrict__ cx,
                             float* __restrict__ enc_s, float* __restrict__ enc_e) {
  int idx = blockIdx.x * blockDim.x + threadIdx.x;   // over Bn * 2H
  if (idx >= Bn * TWOH) return;
  int b = idx / TWOH, c = idx % TWOH;
  float v = enc[((size_t)b * TWOH + c) * Tn + 0];
  enc_s[idx] = v;
  enc_e[idx] = v;
  if (c < Hn) { hx[b * Hn + c] = 0.0f; cx[b * Hn + c] = 0.0f; }
}

// r = tanh(concat(hx, enc_s, enc_e) @ WD^T)   [B, H]
__global__ void k_compute_r(const float* __restrict__ hx,
                            const float* __restrict__ enc_s,
                            const float* __restrict__ enc_e,
                            const float* __restrict__ WD,   // [H, 5H]
                            float* __restrict__ r) {
  int idx = blockIdx.x * blockDim.x + threadIdx.x;
  if (idx >= Bn * Hn) return;
  int b = idx / Hn, h = idx % Hn;
  const float* w = WD + (size_t)h * FIVEH;
  float acc = 0.0f;
  for (int k = 0; k < Hn; ++k)   acc += hx[b * Hn + k]      * w[k];
  for (int k = 0; k < TWOH; ++k) acc += enc_s[b * TWOH + k] * w[Hn + k];
  for (int k = 0; k < TWOH; ++k) acc += enc_e[b * TWOH + k] * w[THREEH + k];
  r[idx] = tanhf(acc);
}

// er[:, :, 512:768] = broadcast r over T
__global__ void k_broadcast_r(const float* __restrict__ r,
                              unsigned short* __restrict__ er) {
  int idx = blockIdx.x * blockDim.x + threadIdx.x;   // over Mrows * H
  if (idx >= Mrows * Hn) return;
  int m = idx / Hn, h = idx % Hn;
  int b = m / Tn;
  er[(size_t)m * THREEH + TWOH + h] = f2bf(r[b * Hn + h]);
}

// ---------------------------------------------------------------------------
// Fused bf16 WMMA GEMM + bias + maxout(P=16).
// Block = 8 waves; tile = 128 M x 64 N. Each wave owns one 16-row M-tile and
// 4 N-groups (register blocking: one A-fragment feeds 4 WMMAs). The 64x32
// W k-slice (4 KB) is staged in LDS (double-buffered) — via the Tensor Data
// Mover when available (D# per ISA §8: 2D tile, data_size=4B, tile 16 dwords
// x 64 rows, row stride K/2 dwords), else cooperative b128 copies.
//
// WMMA wave32 operand layouts (ISA 7.12.2):
//   A 16-bit: lane L holds row M=L%16; element j=2v+s maps to
//             K = (v&3)*2 + s + (L/16)*8 + (v>>2)*16.  B mirrors it (N=L%16).
//   C/D f32:  VGPR i holds row i + 8*(L/16), column L%16.
// ---------------------------------------------------------------------------
__global__ void __launch_bounds__(256)
k_gemm_maxout(const unsigned short* __restrict__ A, int lda,
              const unsigned short* __restrict__ W,   // [Ngroups*16, K]
              const float* __restrict__ bias,
              unsigned short* __restrict__ Out, int ldo,
              int M, int Ngroups, int K) {
  __shared__ __attribute__((aligned(16))) unsigned short wtile[2][64 * 32]; // 2 x 4KB

  const int tid  = threadIdx.x;
  const int lane = tid & 31;
  const int wv   = tid >> 5;            // wave in block, 0..7
  const int half = lane >> 4;
  const int l16  = lane & 15;

  const int nChunks = Ngroups >> 2;     // N-blocks of 4 groups (64 cols)
  const int nBlk = blockIdx.x % nChunks;
  const int mBlk = blockIdx.x / nChunks;
  const int n0   = nBlk * 64;           // first N column of this block
  const int mT   = mBlk * 8 + wv;       // this wave's 16-row M tile
  const int row  = mT * 16 + l16;

  const unsigned short* arow = A + (size_t)row * lda;

  // fallback cooperative staging indices (also used when no TDM)
  const int srow   = tid >> 2;          // 0..63
  const int schunk = tid & 3;           // 16-byte chunk within row
  const unsigned short* wsrc = W + (size_t)(n0 + srow) * K + schunk * 8;
  const int sdst = srow * 32 + schunk * 8;

  const int nsteps     = K >> 5;
  const int rows_total = Ngroups << 4;  // tensor_dim1
  const int Kdw        = K >> 1;        // dwords per W row

  auto stage = [&](int buf, int k0) {
#if ATHENA_USE_TDM
    if (wv == 0) {
      unsigned long long ga =
          (unsigned long long)(uintptr_t)(const void*)(W + (size_t)n0 * K + k0);
      unsigned int lds_off = (unsigned int)(uintptr_t)(void*)&wtile[buf][0];
      u32x4 g0;
      g0[0] = 1u;                                   // count=1, no gather
      g0[1] = lds_off;                              // lds_addr
      g0[2] = (unsigned int)ga;                     // global_addr[31:0]
      g0[3] = (unsigned int)((ga >> 32) & 0x01FFFFFFu) | (2u << 30); // addr hi | type=2
      i32x8 g1;
      g1[0] = (int)(2u << 16);                      // mask=0, data_size=4B
      g1[1] = (int)(((unsigned)Kdw & 0xFFFFu) << 16);        // tensor_dim0 lo16
      g1[2] = (int)(((unsigned)Kdw >> 16) |
                    (((unsigned)rows_total & 0xFFFFu) << 16)); // dim0 hi | dim1 lo
      g1[3] = (int)(((unsigned)rows_total >> 16) | (16u << 16)); // dim1 hi | tile_dim0=16
      g1[4] = (int)64;                              // tile_dim1=64, tile_dim2=0
      g1[5] = (int)Kdw;                             // tensor_dim0_stride lo32
      g1[6] = 0;                                    // stride hi | dim1_stride lo
      g1[7] = 0;
      i32x4 g2 = {0, 0, 0, 0};
      i32x4 g3 = {0, 0, 0, 0};
#if defined(__clang_major__) && (__clang_major__ >= 23)
      i32x8 g4 = {0, 0, 0, 0, 0, 0, 0, 0};
      __builtin_amdgcn_tensor_load_to_lds(g0, g1, g2, g3, g4, 0);
#else
      __builtin_amdgcn_tensor_load_to_lds(g0, g1, g2, g3, 0);
#endif
    }
#else
    uint4 t = *(const uint4*)(wsrc + k0);
    *(uint4*)&wtile[buf][sdst] = t;
#endif
  };

  // prologue: stage first k-slice
  stage(0, 0);
#if ATHENA_USE_TDM
  __builtin_amdgcn_s_wait_tensorcnt(0);
#endif
  __syncthreads();

  v8f zero = {};
  v8f c[4];
  c[0] = zero; c[1] = zero; c[2] = zero; c[3] = zero;

  int cur = 0;
  for (int ks = 0; ks < nsteps; ++ks) {
    const int k0 = ks << 5;
    if (ks + 1 < nsteps) {
      stage(cur ^ 1, k0 + 32);                       // async stage next slice
      __builtin_prefetch(arow + k0 + 32, 0, 3);      // global_prefetch next A
    }

    union { unsigned int u[8]; v16bf v; } av;
#pragma unroll
    for (int v = 0; v < 8; ++v) {
      int kk = k0 + ((v & 3) << 1) + (half << 3) + ((v >> 2) << 4);
      av.u[v] = *(const unsigned int*)(arow + kk);
    }

#pragma unroll
    for (int g = 0; g < 4; ++g) {
      const unsigned short* wr = &wtile[cur][(g * 16 + l16) * 32];
      union { unsigned int u[8]; v16bf v; } bv;
#pragma unroll
      for (int v = 0; v < 8; ++v) {
        int kk = ((v & 3) << 1) + (half << 3) + ((v >> 2) << 4);
        bv.u[v] = *(const unsigned int*)(wr + kk);
      }
      c[g] = __builtin_amdgcn_wmma_f32_16x16x32_bf16(false, av.v, false, bv.v,
                                                     (short)0, c[g], false, false);
    }

#if ATHENA_USE_TDM
    __builtin_amdgcn_s_wait_tensorcnt(0);
#endif
    __syncthreads();
    cur ^= 1;
  }

  // epilogue: bias + maxout over the 16 columns of each group, store bf16
#pragma unroll
  for (int g = 0; g < 4; ++g) {
    float bval = bias[n0 + g * 16 + l16];
    v8f acc = c[g];
#pragma unroll
    for (int i = 0; i < 8; ++i) acc[i] += bval;
#pragma unroll
    for (int m = 1; m < 16; m <<= 1)
#pragma unroll
      for (int i = 0; i < 8; ++i)
        acc[i] = fmaxf(acc[i], __shfl_xor(acc[i], m, 16));
    if (l16 == 0) {
#pragma unroll
      for (int i = 0; i < 8; ++i) {
        int r = mT * 16 + half * 8 + i;
        Out[(size_t)r * ldo + (nBlk * 4 + g)] = f2bf(acc[i]);
      }
    }
  }
}

// GEMM3: [M,K] @ W[16,K]^T, bias, maxout over 16, minus mask -> f32 scores.
__global__ void k_gemm_score(const unsigned short* __restrict__ A, int lda,
                             const unsigned short* __restrict__ W,  // [16, K]
                             const float* __restrict__ bias,        // [16]
                             const int* __restrict__ plens,
                             float* __restrict__ out,               // [B*T]
                             int K) {
  int lane = threadIdx.x & 31;
  int wave = (blockIdx.x * blockDim.x + threadIdx.x) >> 5;
  if (wave >= (Mrows >> 4)) return;
  int mT = wave;
  int half = lane >> 4;
  int l16  = lane & 15;

  const unsigned short* arow = A + (size_t)(mT * 16 + l16) * lda;
  const unsigned short* wrow = W + (size_t)l16 * K;

  v8f c = {};
  for (int k0 = 0; k0 < K; k0 += 32) {
    union { unsigned int u[8]; v16bf v; } av, bv;
#pragma unroll
    for (int v = 0; v < 8; ++v) {
      int kk = k0 + ((v & 3) << 1) + (half << 3) + ((v >> 2) << 4);
      av.u[v] = *(const unsigned int*)(arow + kk);
      bv.u[v] = *(const unsigned int*)(wrow + kk);
    }
    c = __builtin_amdgcn_wmma_f32_16x16x32_bf16(false, av.v, false, bv.v,
                                                (short)0, c, false, false);
  }

  float bval = bias[l16];
#pragma unroll
  for (int i = 0; i < 8; ++i) c[i] += bval;
#pragma unroll
  for (int m = 1; m < 16; m <<= 1)
#pragma unroll
    for (int i = 0; i < 8; ++i)
      c[i] = fmaxf(c[i], __shfl_xor(c[i], m, 16));

  if (l16 == 0) {
#pragma unroll
    for (int i = 0; i < 8; ++i) {
      int row = mT * 16 + half * 8 + i;
      int b = row / Tn, t = row % Tn;
      float mask = (t < plens[b]) ? 0.0f : MASK_PENALTY;
      out[row] = c[i] - mask;
    }
  }
}

// argmax over T (first-max tie-break) + gather selected encoder column.
__global__ void k_argmax_gather(const float* __restrict__ scores,   // [B*T]
                                const float* __restrict__ enc,      // [B,2H,T]
                                int* __restrict__ idx_out,          // [B]
                                float* __restrict__ enc_sel) {      // [B,2H]
  int b = blockIdx.x;
  int lane = threadIdx.x;
  float best = -3.4e38f;
  int   bi   = 0;
  for (int t = lane; t < Tn; t += 32) {
    float v = scores[b * Tn + t];
    if (v > best) { best = v; bi = t; }
  }
#pragma unroll
  for (int m = 16; m >= 1; m >>= 1) {
    float ov = __shfl_xor(best, m, 32);
    int   oi = __shfl_xor(bi,   m, 32);
    if (ov > best || (ov == best && oi < bi)) { best = ov; bi = oi; }
  }
  if (lane == 0) idx_out[b] = bi;
  for (int c = lane; c < TWOH; c += 32)
    enc_sel[b * TWOH + c] = enc[((size_t)b * TWOH + c) * Tn + bi];
}

// LSTM gates: gates[b,j] = x @ W_ih^T + b_ih + hx @ W_hh^T + b_hh
__global__ void k_lstm_gates(const float* __restrict__ enc_s,
                             const float* __restrict__ enc_e,
                             const float* __restrict__ hx,
                             const float* __restrict__ W_ih,  // [4H,4H]
                             const float* __restrict__ W_hh,  // [4H,H]
                             const float* __restrict__ b_ih,
                             const float* __restrict__ b_hh,
                             float* __restrict__ gates) {
  int idx = blockIdx.x * blockDim.x + threadIdx.x;
  if (idx >= Bn * FOURH) return;
  int b = idx / FOURH, j = idx % FOURH;
  const float* wi = W_ih + (size_t)j * FOURH;
  const float* wh = W_hh + (size_t)j * Hn;
  float g = b_ih[j] + b_hh[j];
  for (int k = 0; k < TWOH; ++k) g += enc_s[b * TWOH + k] * wi[k];
  for (int k = 0; k < TWOH; ++k) g += enc_e[b * TWOH + k] * wi[TWOH + k];
  for (int k = 0; k < Hn;   ++k) g += hx[b * Hn + k] * wh[k];
  gates[idx] = g;
}

static __device__ __forceinline__ float sigm(float x) { return 1.0f / (1.0f + expf(-x)); }

__global__ void k_lstm_update(const float* __restrict__ gates,
                              float* __restrict__ cx,
                              float* __restrict__ hcell) {
  int idx = blockIdx.x * blockDim.x + threadIdx.x;
  if (idx >= Bn * Hn) return;
  int b = idx / Hn, h = idx % Hn;
  const float* g = gates + (size_t)b * FOURH;
  float i_ = g[h], f_ = g[Hn + h], g_ = g[2 * Hn + h], o_ = g[3 * Hn + h];
  float c = sigm(f_) * cx[idx] + sigm(i_) * tanhf(g_);
  cx[idx] = c;
  hcell[idx] = sigm(o_) * tanhf(c);
}

__global__ void k_mlp(const float* __restrict__ hcell,
                      const float* __restrict__ W_mlp,  // [H,H]
                      const float* __restrict__ b_mlp,
                      float* __restrict__ hx) {
  int idx = blockIdx.x * blockDim.x + threadIdx.x;
  if (idx >= Bn * Hn) return;
  int b = idx / Hn, h = idx % Hn;
  const float* w = W_mlp + (size_t)h * Hn;
  float acc = b_mlp[h];
  for (int k = 0; k < Hn; ++k) acc += hcell[b * Hn + k] * w[k];
  hx[idx] = acc;
}

__global__ void k_finalize(const int* __restrict__ start_i,
                           const int* __restrict__ end_i,
                           float* __restrict__ out) {
  int i = threadIdx.x;
  if (i < Bn)           out[i] = (float)start_i[i];
  else if (i < 2 * Bn)  out[i] = (float)end_i[i - Bn];
}

// ---------------------------------------------------------------------------
extern "C" void kernel_launch(void* const* d_in, const int* in_sizes, int n_in,
                              void* d_out, int out_size, void* d_ws, size_t ws_size,
                              hipStream_t stream) {
  (void)in_sizes; (void)n_in; (void)out_size; (void)ws_size;

  const float* enc   = (const float*)d_in[0];
  const int*   plens = (const int*)d_in[1];
  const float* WD_s  = (const float*)d_in[2];
  const float* w1s   = (const float*)d_in[3];
  const float* b1s   = (const float*)d_in[4];
  const float* w2s   = (const float*)d_in[5];
  const float* b2s   = (const float*)d_in[6];
  const float* w3s   = (const float*)d_in[7];
  const float* b3s   = (const float*)d_in[8];
  const float* WD_e  = (const float*)d_in[9];
  const float* w1e   = (const float*)d_in[10];
  const float* b1e   = (const float*)d_in[11];
  const float* w2e   = (const float*)d_in[12];
  const float* b2e   = (const float*)d_in[13];
  const float* w3e   = (const float*)d_in[14];
  const float* b3e   = (const float*)d_in[15];
  const float* W_ih  = (const float*)d_in[16];
  const float* W_hh  = (const float*)d_in[17];
  const float* b_ih  = (const float*)d_in[18];
  const float* b_hh  = (const float*)d_in[19];
  const float* W_mlp = (const float*)d_in[20];
  const float* b_mlp = (const float*)d_in[21];

  float* out = (float*)d_out;
  float* alphas = out + 2 * Bn;                     // [ITERS, B*T]
  float* betas  = alphas + NITERS * Mrows;          // [ITERS, B*T]

  // ---- workspace carve-out ----
  char* base = (char*)d_ws;
  size_t off = 0;
  auto carve = [&](size_t bytes) -> char* {
    char* p = base + off;
    off = (off + bytes + 255) & ~(size_t)255;
    return p;
  };
  unsigned short* er     = (unsigned short*)carve((size_t)Mrows * THREEH * 2); // 24 MB
  unsigned short* m12    = (unsigned short*)carve((size_t)Mrows * TWOH * 2);   // 16 MB
  unsigned short* w1s_b  = (unsigned short*)carve((size_t)HP * THREEH * 2);
  unsigned short* w2s_b  = (unsigned short*)carve((size_t)HP * Hn * 2);
  unsigned short* w3s_b  = (unsigned short*)carve((size_t)Pn * TWOH * 2);
  unsigned short* w1e_b  = (unsigned short*)carve((size_t)HP * THREEH * 2);
  unsigned short* w2e_b  = (unsigned short*)carve((size_t)HP * Hn * 2);
  unsigned short* w3e_b  = (unsigned short*)carve((size_t)Pn * TWOH * 2);
  float* r_f   = (float*)carve((size_t)Bn * Hn * 4);
  float* hx    = (float*)carve((size_t)Bn * Hn * 4);
  float* cx    = (float*)carve((size_t)Bn * Hn * 4);
  float* hcell = (float*)carve((size_t)Bn * Hn * 4);
  float* gates = (float*)carve((size_t)Bn * FOURH * 4);
  float* enc_s = (float*)carve((size_t)Bn * TWOH * 4);
  float* enc_e = (float*)carve((size_t)Bn * TWOH * 4);
  int* start_i = (int*)carve((size_t)Bn * 4);
  int* end_i   = (int*)carve((size_t)Bn * 4);

  const int BT = 256;
  auto blocks = [](int n, int bt) { return (n + bt - 1) / bt; };

  // ---- one-time conversions ----
  k_f32_to_bf16<<<blocks(HP * THREEH, BT), BT, 0, stream>>>(w1s, w1s_b, HP * THREEH);
  k_f32_to_bf16<<<blocks(HP * Hn, BT),    BT, 0, stream>>>(w2s, w2s_b, HP * Hn);
  k_f32_to_bf16<<<blocks(Pn * TWOH, BT),  BT, 0, stream>>>(w3s, w3s_b, Pn * TWOH);
  k_f32_to_bf16<<<blocks(HP * THREEH, BT), BT, 0, stream>>>(w1e, w1e_b, HP * THREEH);
  k_f32_to_bf16<<<blocks(HP * Hn, BT),    BT, 0, stream>>>(w2e, w2e_b, HP * Hn);
  k_f32_to_bf16<<<blocks(Pn * TWOH, BT),  BT, 0, stream>>>(w3e, w3e_b, Pn * TWOH);
  k_build_er_base<<<blocks(Mrows * TWOH, BT), BT, 0, stream>>>(enc, er);
  k_init_state<<<blocks(Bn * TWOH, BT), BT, 0, stream>>>(enc, hx, cx, enc_s, enc_e);

  const int gemmBlocks  = (Mrows / 128) * (Hn / 4);   // 128 x 64 = 8192 blocks
  const int scoreTiles  = Mrows / 16;
  const int scoreBlocks = (scoreTiles + (BT / 32) - 1) / (BT / 32);

  for (int it = 0; it < NITERS; ++it) {
    // ---------- start pointer ----------
    k_compute_r<<<blocks(Bn * Hn, BT), BT, 0, stream>>>(hx, enc_s, enc_e, WD_s, r_f);
    k_broadcast_r<<<blocks(Mrows * Hn, BT), BT, 0, stream>>>(r_f, er);
    k_gemm_maxout<<<gemmBlocks, BT, 0, stream>>>(er, THREEH, w1s_b, b1s,
                                                 m12, TWOH, Mrows, Hn, THREEH);
    k_gemm_maxout<<<gemmBlocks, BT, 0, stream>>>(m12, TWOH, w2s_b, b2s,
                                                 m12 + Hn, TWOH, Mrows, Hn, Hn);
    k_gemm_score<<<scoreBlocks, BT, 0, stream>>>(m12, TWOH, w3s_b, b3s, plens,
                                                 alphas + (size_t)it * Mrows, TWOH);
    k_argmax_gather<<<Bn, 32, 0, stream>>>(alphas + (size_t)it * Mrows, enc,
                                           start_i, enc_s);

    // ---------- end pointer ----------
    k_compute_r<<<blocks(Bn * Hn, BT), BT, 0, stream>>>(hx, enc_s, enc_e, WD_e, r_f);
    k_broadcast_r<<<blocks(Mrows * Hn, BT), BT, 0, stream>>>(r_f, er);
    k_gemm_maxout<<<gemmBlocks, BT, 0, stream>>>(er, THREEH, w1e_b, b1e,
                                                 m12, TWOH, Mrows, Hn, THREEH);
    k_gemm_maxout<<<gemmBlocks, BT, 0, stream>>>(m12, TWOH, w2e_b, b2e,
                                                 m12 + Hn, TWOH, Mrows, Hn, Hn);
    k_gemm_score<<<scoreBlocks, BT, 0, stream>>>(m12, TWOH, w3e_b, b3e, plens,
                                                 betas + (size_t)it * Mrows, TWOH);
    k_argmax_gather<<<Bn, 32, 0, stream>>>(betas + (size_t)it * Mrows, enc,
                                           end_i, enc_e);

    // ---------- LSTM cell + MLP ----------
    k_lstm_gates<<<blocks(Bn * FOURH, BT), BT, 0, stream>>>(enc_s, enc_e, hx,
                                                            W_ih, W_hh, b_ih, b_hh, gates);
    k_lstm_update<<<blocks(Bn * Hn, BT), BT, 0, stream>>>(gates, cx, hcell);
    k_mlp<<<blocks(Bn * Hn, BT), BT, 0, stream>>>(hcell, W_mlp, b_mlp, hx);
  }

  k_finalize<<<1, 64, 0, stream>>>(start_i, end_i, out);
}
